// Sigmoid_Attention_51402168598902
// MI455X (gfx1250) — matrix operand order repaired and checked by reading, not compiled, not run on hardware
//
#include <hip/hip_runtime.h>
#include <stdint.h>

// ---------------- problem constants ----------------
#define B_   32
#define N_   577
#define C_   768
#define H_   12
#define DH_  64
#define SCALE_ 0.125f   // 64^-0.5

typedef unsigned short u16;
typedef __attribute__((ext_vector_type(16))) __bf16 v16bf;
typedef __attribute__((ext_vector_type(8)))  float  v8f;

struct U4x2 { uint4 a, b; };

// Load a 16-element bf16 fragment as two 16-byte chunks.
__device__ __forceinline__ v16bf ld_frag(const u16* p0, const u16* p1) {
  U4x2 u;
  u.a = *reinterpret_cast<const uint4*>(p0);
  u.b = *reinterpret_cast<const uint4*>(p1);
  return __builtin_bit_cast(v16bf, u);
}

__device__ __forceinline__ u16 f2bf(float f) {  // round-to-nearest-even
  uint32_t u = __builtin_bit_cast(uint32_t, f);
  u += 0x7FFFu + ((u >> 16) & 1u);
  return (u16)(u >> 16);
}

__device__ __forceinline__ v8f wmma_bf16(v16bf a, v16bf b, v8f c) {
  // D = A(16x32 bf16) * B(32x16 bf16) + C(16x16 f32)
  return __builtin_amdgcn_wmma_f32_16x16x32_bf16(false, a, false, b,
                                                 (short)0, c, false, false);
}

__device__ __forceinline__ void wavefence() {
#if __has_builtin(__builtin_amdgcn_wave_barrier)
  __builtin_amdgcn_wave_barrier();
#else
  asm volatile("" ::: "memory");
#endif
}

// ---- CDNA5 async global->LDS copy (ASYNCcnt path, §08) ----
// Each lane copies 16B from its global address to its LDS address.
__device__ __forceinline__ void async_ld_b128(void* lds, const void* g) {
  asm volatile("global_load_async_to_lds_b128 %0, %1, off"
               :: "v"((uint32_t)(uintptr_t)lds), "v"((uint64_t)(uintptr_t)g)
               : "memory");
}
__device__ __forceinline__ void wait_async0() {
  asm volatile("s_wait_asynccnt 0x0" ::: "memory");
}

// ---------------- fp32 -> bf16 convert ----------------
__global__ void cvt_bf16_kernel(const float* __restrict__ in,
                                u16* __restrict__ out, long n) {
  long i = (long)blockIdx.x * blockDim.x + threadIdx.x;
  long stride = (long)gridDim.x * blockDim.x;
  for (; i < n; i += stride) out[i] = f2bf(in[i]);
}

// ---------------- bf16 GEMM: C[M,Ncols] = A[M,K] * Bw[Ncols,K]^T + bias ----------------
// 256 threads (8 waves), 128x128 tile, async-staged LDS tiles (k-step 32).
// Wave (wm,wn) in 2x4 grid computes a 64x32 subtile = 4x2 wmma frags.
// Requires Ncols % 128 == 0 (holds: 2304, 768).
template <int OUT_BF16>
__global__ void gemm_bf16_kernel(const u16* __restrict__ A,
                                 const u16* __restrict__ Bw,
                                 const float* __restrict__ bias,
                                 void* __restrict__ Cout,
                                 int M, int Ncols, int K) {
  __shared__ u16 At[128 * 32];
  __shared__ u16 Bt[128 * 32];

  const int tid  = threadIdx.x;
  const int lane = tid & 31;
  const int wave = tid >> 5;
  const int lrow = lane & 15;
  const int half = lane >> 4;
  const int wm = wave >> 2;          // 0..1
  const int wn = wave & 3;           // 0..3
  const int m0 = blockIdx.x * 128;
  const int n0 = blockIdx.y * 128;
  const int mw = wm * 64;
  const int nw = wn * 32;

  v8f acc[4][2] = {};

  for (int k0 = 0; k0 < K; k0 += 32) {
    // ---- async stage A/B tiles: 128 rows x 32 cols bf16 each ----
    for (int c = tid; c < 512; c += 256) {
      int row = c >> 2, ch = (c & 3) * 8;
      int ar = m0 + row; if (ar > M - 1) ar = M - 1;
      async_ld_b128(&At[row * 32 + ch], A  + (size_t)ar * K        + k0 + ch);
      async_ld_b128(&Bt[row * 32 + ch], Bw + (size_t)(n0 + row) * K + k0 + ch);
    }
    wait_async0();
    __syncthreads();

    v16bf af[4], bfr[2];
#pragma unroll
    for (int i = 0; i < 4; i++) {
      const u16* p = At + (mw + i * 16 + lrow) * 32 + half * 8;
      af[i] = ld_frag(p, p + 16);
    }
#pragma unroll
    for (int j = 0; j < 2; j++) {
      const u16* p = Bt + (nw + j * 16 + lrow) * 32 + half * 8;
      bfr[j] = ld_frag(p, p + 16);
    }
#pragma unroll
    for (int i = 0; i < 4; i++)
#pragma unroll
      for (int j = 0; j < 2; j++)
        acc[i][j] = wmma_bf16(af[i], bfr[j], acc[i][j]);
    __syncthreads();  // tile reads done before restage
  }

#pragma unroll
  for (int j = 0; j < 2; j++) {
    int col = n0 + nw + j * 16 + lrow;
    float bv = bias[col];
#pragma unroll
    for (int i = 0; i < 4; i++) {
#pragma unroll
      for (int e = 0; e < 8; e++) {
        int row = m0 + mw + i * 16 + (half ? e + 8 : e);
        if (row < M) {
          float v = acc[i][j][e] + bv;
          if (OUT_BF16)
            ((u16*)Cout)[(size_t)row * Ncols + col] = f2bf(v);
          else
            ((float*)Cout)[(size_t)row * Ncols + col] = v;
        }
      }
    }
  }
}

// ---------------- fused flash attention ----------------
// grid.x = B*H, grid.y = ceil(N/64); block = 128 (4 waves, one 16-row Q tile each)
// qkv layout: row r=(b*N+n), cols [0,C)=Q, [C,2C)=K, [2C,3C)=V, head offset h*DH
__global__ void attn_kernel(const u16* __restrict__ qkv, u16* __restrict__ ob) {
  __shared__ u16 Kt[32 * DH_];       // K tile row-major: [key][d]   (async staged)
  __shared__ u16 Vt[DH_ * 32];       // V tile transposed: [d][key]
  __shared__ u16 Pt[4][16 * 32];     // per-wave P staging, row-major [16][32]

  const int lane = threadIdx.x & 31;
  const int wave = threadIdx.x >> 5;
  const int lrow = lane & 15;
  const int half = lane >> 4;

  const int bi = blockIdx.x / H_;
  const int h  = blockIdx.x % H_;
  const int n0 = blockIdx.y * 64 + wave * 16;

  const size_t rstride = 3 * (size_t)C_;
  const size_t rbase   = (size_t)bi * N_;

  // Q fragments (A operand), rows clamped
  int qr = n0 + lrow; if (qr > N_ - 1) qr = N_ - 1;
  const u16* qp = qkv + (rbase + qr) * rstride + h * DH_;
  v16bf qf[2];
  qf[0] = ld_frag(qp + half * 8,      qp + half * 8 + 16);
  qf[1] = ld_frag(qp + 32 + half * 8, qp + 32 + half * 8 + 16);

  float m_[8], l_[8];
#pragma unroll
  for (int e = 0; e < 8; e++) { m_[e] = -INFINITY; l_[e] = 0.0f; }
  v8f o[4] = {};

  u16* P = &Pt[wave][0];

  for (int kb = 0; kb < N_; kb += 32) {
    // ---- stage K tile (async -> LDS) and V tile (transposed, manual) ----
    for (int c = threadIdx.x; c < 256; c += 128) {
      int row = c >> 3, ch = (c & 7) * 8;
      int kr = kb + row; if (kr > N_ - 1) kr = N_ - 1;
      const u16* grow = qkv + (rbase + kr) * rstride + h * DH_;
      async_ld_b128(&Kt[row * DH_ + ch], grow + C_ + ch);
      uint4 vv = *reinterpret_cast<const uint4*>(grow + 2 * C_ + ch);
      const u16* pv = reinterpret_cast<const u16*>(&vv);
#pragma unroll
      for (int i = 0; i < 8; i++) Vt[(ch + i) * 32 + row] = pv[i];
    }
    wait_async0();
    __syncthreads();

    // ---- S = (Q * SCALE) @ K^T for 32 keys (two 16-key blocks), K from LDS ----
    v8f s[2] = {};
#pragma unroll
    for (int blk = 0; blk < 2; blk++) {
      const u16* kp = Kt + (blk * 16 + lrow) * DH_;
      v16bf b0 = ld_frag(kp + half * 8,      kp + half * 8 + 16);
      v16bf b1 = ld_frag(kp + 32 + half * 8, kp + 32 + half * 8 + 16);
      s[blk] = wmma_bf16(qf[0], b0, s[blk]);
      s[blk] = wmma_bf16(qf[1], b1, s[blk]);
    }

    // scale + mask out-of-range key columns
    const bool v0 = (kb + lrow) < N_;
    const bool v1 = (kb + 16 + lrow) < N_;
#pragma unroll
    for (int e = 0; e < 8; e++) {
      s[0][e] = v0 ? s[0][e] * SCALE_ : -INFINITY;
      s[1][e] = v1 ? s[1][e] * SCALE_ : -INFINITY;
    }

    // ---- online softmax (row reductions across 16-lane halves) ----
    float mnew[8], alpha[8];
#pragma unroll
    for (int e = 0; e < 8; e++) {
      float t = fmaxf(s[0][e], s[1][e]);
      t = fmaxf(t, __shfl_xor(t, 1, 32));
      t = fmaxf(t, __shfl_xor(t, 2, 32));
      t = fmaxf(t, __shfl_xor(t, 4, 32));
      t = fmaxf(t, __shfl_xor(t, 8, 32));
      mnew[e]  = fmaxf(m_[e], t);
      alpha[e] = __expf(m_[e] - mnew[e]);
    }
#pragma unroll
    for (int e = 0; e < 8; e++) {
      s[0][e] = __expf(s[0][e] - mnew[e]);
      s[1][e] = __expf(s[1][e] - mnew[e]);
      float t = s[0][e] + s[1][e];
      t += __shfl_xor(t, 1, 32);
      t += __shfl_xor(t, 2, 32);
      t += __shfl_xor(t, 4, 32);
      t += __shfl_xor(t, 8, 32);
      l_[e] = l_[e] * alpha[e] + t;
      m_[e] = mnew[e];
    }
#pragma unroll
    for (int a = 0; a < 4; a++)
#pragma unroll
      for (int e = 0; e < 8; e++) o[a][e] *= alpha[e];

    // ---- stage P (C-layout -> row-major LDS), re-read as A fragment ----
#pragma unroll
    for (int e = 0; e < 8; e++) {
      int r = half ? e + 8 : e;
      P[r * 32 + lrow]      = f2bf(s[0][e]);
      P[r * 32 + 16 + lrow] = f2bf(s[1][e]);
    }
    wavefence();  // same-wave DS ops are in-order; block compiler reordering
    const u16* pp = P + lrow * 32 + half * 8;
    v16bf pf = ld_frag(pp, pp + 16);

    // ---- O += P @ V  (4 d-tiles of 16) ----
#pragma unroll
    for (int a = 0; a < 4; a++) {
      const u16* vp = Vt + (a * 16 + lrow) * 32 + half * 8;
      v16bf vf = ld_frag(vp, vp + 16);
      o[a] = wmma_bf16(pf, vf, o[a]);
    }
    __syncthreads();  // Kt/Vt reads done before next tile restages
  }

  // ---- normalize + store attention output as bf16 [B,N,C] ----
#pragma unroll
  for (int a = 0; a < 4; a++)
#pragma unroll
    for (int e = 0; e < 8; e++) {
      int n = n0 + (half ? e + 8 : e);
      if (n < N_) {
        float val = o[a][e] / l_[e];
        ob[(rbase + n) * C_ + h * DH_ + a * 16 + lrow] = f2bf(val);
      }
    }
}

// ---------------- launch ----------------
extern "C" void kernel_launch(void* const* d_in, const int* in_sizes, int n_in,
                              void* d_out, int out_size, void* d_ws, size_t ws_size,
                              hipStream_t stream) {
  (void)in_sizes; (void)n_in; (void)out_size; (void)ws_size;

  const float* x      = (const float*)d_in[0];
  const float* qkv_w  = (const float*)d_in[1];
  const float* qkv_b  = (const float*)d_in[2];
  const float* proj_w = (const float*)d_in[3];
  const float* proj_b = (const float*)d_in[4];
  float* out = (float*)d_out;

  const size_t BN = (size_t)B_ * N_;               // 18464
  char* ws = (char*)d_ws;
  size_t off = 0;
  auto take = [&](size_t bytes) { char* p = ws + off; off = (off + bytes + 255) & ~(size_t)255; return p; };
  u16* xb    = (u16*)take(BN * C_ * 2);            // x in bf16
  u16* wqkvb = (u16*)take((size_t)3 * C_ * C_ * 2);// qkv_w bf16
  u16* wprjb = (u16*)take((size_t)C_ * C_ * 2);    // proj_w bf16
  u16* qkvb  = (u16*)take(BN * 3 * C_ * 2);        // qkv activations bf16
  u16* ob    = (u16*)take(BN * C_ * 2);            // attention output bf16

  // 1) fp32 -> bf16 conversions
  cvt_bf16_kernel<<<4096, 256, 0, stream>>>(x, xb, (long)(BN * C_));
  cvt_bf16_kernel<<<2048, 256, 0, stream>>>(qkv_w, wqkvb, (long)3 * C_ * C_);
  cvt_bf16_kernel<<<1024, 256, 0, stream>>>(proj_w, wprjb, (long)C_ * C_);

  // 2) QKV GEMM: [BN,768] x [2304,768]^T -> bf16 [BN,2304]
  {
    dim3 grid((unsigned)((BN + 127) / 128), (unsigned)(3 * C_ / 128));
    gemm_bf16_kernel<1><<<grid, 256, 0, stream>>>(xb, wqkvb, qkv_b, (void*)qkvb,
                                                  (int)BN, 3 * C_, C_);
  }

  // 3) fused flash attention -> bf16 [B,N,C]
  {
    dim3 grid(B_ * H_, (N_ + 63) / 64);
    attn_kernel<<<grid, 128, 0, stream>>>(qkvb, ob);
  }

  // 4) proj GEMM: [BN,768] x [768,768]^T + bias -> fp32 output
  {
    dim3 grid((unsigned)((BN + 127) / 128), (unsigned)(C_ / 128));
    gemm_bf16_kernel<0><<<grid, 256, 0, stream>>>(ob, wprjb, proj_b, (void*)out,
                                                  (int)BN, C_, C_);
  }
}
